// ResidualCodebookCollection_39170101739980
// MI455X (gfx1250) — matrix operand order repaired and checked by reading, not compiled, not run on hardware
//
#include <hip/hip_runtime.h>
#include <hip/hip_bf16.h>

typedef __attribute__((ext_vector_type(16))) __bf16 v16bf;
typedef __attribute__((ext_vector_type(8)))  float  v8f;

#define B_   16
#define N_   2048
#define D_   256
#define C_   8
#define K_   1024
#define NTOK (B_ * N_)          // 32768 tokens
#define ROWF 260                // xres/zq LDS row stride (floats): banks 4 apart
#define TILEF (16 * ROWF)

// ---- LDS layout (bytes from dynamic-LDS base; no static __shared__ anywhere) ----
#define BROW_BF   264                          // B-stage row stride in bf16 (528B = 132 dw)
#define BBUF_BYTES (16 * BROW_BF * 2)          // 8448 B per buffer
#define BSTAGE0   0
#define BSTAGE1   BBUF_BYTES
#define WAVE_BASE 17408                        // 2*8448=16896, padded to 256B multiple
#define WAVE_BYTES ((2 * TILEF + 16) * 4)      // xres + zq + idx16 = 33344 B
#define SMEM_TOTAL (WAVE_BASE + 4 * WAVE_BYTES)  // 150784 B

union BF16x16 { v16bf v; unsigned short u[16]; };

__device__ __forceinline__ unsigned short f2bf(float f) {
    unsigned int x = __float_as_uint(f);
    x += 0x7FFFu + ((x >> 16) & 1u);
    return (unsigned short)(x >> 16);
}

// ---------------------------------------------------------------------------
// Pass 1: codebooks fp32 (C,K,D) -> bf16 copy + 0.5*|e|^2 per entry.
// ---------------------------------------------------------------------------
__global__ void vq_prep(const float* __restrict__ cb,
                        unsigned short* __restrict__ cb_bf,
                        float* __restrict__ enorm05) {
    const int row  = blockIdx.x;
    const int lane = threadIdx.x;
    const float* src = cb + (size_t)row * D_;

    float s = 0.f;
    unsigned short t[8];
#pragma unroll
    for (int j = 0; j < 8; ++j) {
        float v = src[lane * 8 + j];
        s += v * v;
        t[j] = f2bf(v);
    }
    uint4 pk;
    pk.x = (unsigned)t[0] | ((unsigned)t[1] << 16);
    pk.y = (unsigned)t[2] | ((unsigned)t[3] << 16);
    pk.z = (unsigned)t[4] | ((unsigned)t[5] << 16);
    pk.w = (unsigned)t[6] | ((unsigned)t[7] << 16);
    *(uint4*)(cb_bf + (size_t)row * D_ + lane * 8) = pk;

#pragma unroll
    for (int mask = 16; mask >= 1; mask >>= 1)
        s += __shfl_xor(s, mask, 32);
    if (lane == 0) enorm05[row] = 0.5f * s;
}

// ---------------------------------------------------------------------------
// Async-stage one 16-entry x 256-dim bf16 B-tile (8 KB) into LDS, 128 threads,
// 4 x global_load_async_to_lds_b128 per thread (ASYNCcnt path).
// ---------------------------------------------------------------------------
__device__ __forceinline__ void stage_chunk(const unsigned short* cbb,
                                            int entry0, unsigned bufbase,
                                            int tid) {
#pragma unroll
    for (int j = 0; j < 4; ++j) {
        int i   = tid + 128 * j;            // 0..511
        int row = i >> 5;                   // 0..15
        int seg = i & 31;                   // 16B segment within 512B row
        unsigned voff = (unsigned)((entry0 + row) * (D_ * 2) + seg * 16);
        unsigned lds  = bufbase + (unsigned)(row * (BROW_BF * 2) + seg * 16);
        asm volatile("global_load_async_to_lds_b128 %0, %1, %2 offset:0"
                     :: "v"(lds), "v"(voff), "s"(cbb) : "memory");
    }
}

// ---------------------------------------------------------------------------
// Pass 2: residual VQ. 4 waves/block; wave w owns 16 tokens. All waves share
// the double-buffered B-stage in LDS (4x L2 traffic reduction).
// ---------------------------------------------------------------------------
__global__ void __launch_bounds__(128)
vq_main(const float* __restrict__ x_in,
        const float* __restrict__ cb,                // fp32 (C,K,D)
        const unsigned short* __restrict__ cb_bf,
        const float* __restrict__ enorm05,
        float* __restrict__ out_z,                   // (NTOK, D)
        float* __restrict__ out_idx)                 // (B, C, N) as float
{
    extern __shared__ char smem[];
    const int tid  = threadIdx.x;
    const int lane = tid & 31;
    const int wid  = tid >> 5;

    float* xres  = (float*)(smem + WAVE_BASE + wid * WAVE_BYTES);
    float* zq    = xres + TILEF;
    int*   idx16 = (int*)(zq + TILEF);

    const int tile = blockIdx.x * 4 + wid;           // this wave's 16-token tile
    const int tok0 = tile * 16;
    const int m16  = lane & 15;
    const int half = lane >> 4;

    // ---- init: xres = x_in, zq = 0
#pragma unroll 8
    for (int j = 0; j < 128; ++j) {
        int e = j * 32 + lane;
        int m = e >> 8, d = e & 255;
        xres[m * ROWF + d] = x_in[(size_t)(tok0 + m) * D_ + d];
        zq[m * ROWF + d]   = 0.f;
    }
    __syncthreads();

    for (int c = 0; c < C_; ++c) {
        const unsigned short* cbb = cb_bf + (size_t)c * K_ * D_;
        const float*          en  = enorm05 + c * K_;

        // ---- (a) xres -= cumulative zq (faithful to reference recurrence)
#pragma unroll 8
        for (int j = 0; j < 128; ++j) {
            int e = j * 32 + lane;
            int m = e >> 8, d = e & 255;
            xres[m * ROWF + d] -= zq[m * ROWF + d];
        }
        __syncthreads();

        // ---- (b) A fragments: 16x32 bf16 per k-step, contiguous k = 16*half + s,
        //          negated so WMMA accumulates 0.5|e|^2 - x.e
        BF16x16 afrag[8];
#pragma unroll
        for (int ks = 0; ks < 8; ++ks) {
            const float* r = xres + m16 * ROWF + ks * 32 + half * 16;
#pragma unroll
            for (int j = 0; j < 8; ++j) afrag[ks].u[j]     = f2bf(-r[j]);
#pragma unroll
            for (int j = 0; j < 8; ++j) afrag[ks].u[8 + j] = f2bf(-r[8 + j]);
        }

        float bestv[8];
        int   besti[8];
#pragma unroll
        for (int r = 0; r < 8; ++r) { bestv[r] = 3.4e38f; besti[r] = 0; }

        // ---- (c) 64 chunks, double-buffered async B staging
        stage_chunk(cbb, 0, BSTAGE0, tid);
        for (int chunk = 0; chunk < 64; ++chunk) {
            const unsigned bufbase = (chunk & 1) ? BSTAGE1 : BSTAGE0;
            if (chunk < 63) {
                stage_chunk(cbb, (chunk + 1) * 16,
                            ((chunk + 1) & 1) ? BSTAGE1 : BSTAGE0, tid);
                asm volatile("s_wait_asynccnt 0x4" ::: "memory");  // older batch done
            } else {
                asm volatile("s_wait_asynccnt 0x0" ::: "memory");
            }
            __syncthreads();                 // staged tile visible to all waves

            const int entry = chunk * 16 + m16;        // this lane's column
            const float bias = en[entry];
            v8f acc;
#pragma unroll
            for (int r = 0; r < 8; ++r) acc[r] = bias;

            const char* bp = smem + bufbase + m16 * (BROW_BF * 2) + half * 32;
#pragma unroll
            for (int ks = 0; ks < 8; ++ks) {
                v16bf bfrag = *(const v16bf*)(bp + ks * 64);
                acc = __builtin_amdgcn_wmma_f32_16x16x32_bf16(
                        false, afrag[ks].v, false, bfrag,
                        (short)0, acc, false, false);
            }
#pragma unroll
            for (int r = 0; r < 8; ++r) {
                float v = acc[r];
                if (v < bestv[r]) { bestv[r] = v; besti[r] = entry; }
            }
            __syncthreads();                 // done reading before re-stage
        }

        // ---- (d) per-token argmin across the 16 lanes of each half
#pragma unroll
        for (int r = 0; r < 8; ++r) {
            float bv = bestv[r];
            int   bi = besti[r];
#pragma unroll
            for (int mask = 8; mask >= 1; mask >>= 1) {
                float ov = __shfl_xor(bv, mask, 32);
                int   oi = __shfl_xor(bi, mask, 32);
                if (ov < bv || (ov == bv && oi < bi)) { bv = ov; bi = oi; }
            }
            if (m16 == 0) {
                int m = r + 8 * half;
                idx16[m] = bi;
                int tk = tok0 + m;
                int bb = tk >> 11;
                int nn = tk & (N_ - 1);
                out_idx[((size_t)bb * C_ + c) * N_ + nn] = (float)bi;
            }
        }
        __syncthreads();

        // ---- (e) zq += fp32 gather of selected codes
#pragma unroll 8
        for (int j = 0; j < 128; ++j) {
            int e = j * 32 + lane;
            int m = e >> 8, d = e & 255;
            zq[m * ROWF + d] += cb[((size_t)c * K_ + idx16[m]) * D_ + d];
        }
        __syncthreads();
    }

    // ---- write z_q
#pragma unroll 8
    for (int j = 0; j < 128; ++j) {
        int e = j * 32 + lane;
        int m = e >> 8, d = e & 255;
        out_z[(size_t)(tok0 + m) * D_ + d] = zq[m * ROWF + d];
    }
}

// ---------------------------------------------------------------------------
extern "C" void kernel_launch(void* const* d_in, const int* in_sizes, int n_in,
                              void* d_out, int out_size, void* d_ws, size_t ws_size,
                              hipStream_t stream) {
    const float* x_in = (const float*)d_in[0];
    const float* cb   = (const float*)d_in[1];

    unsigned short* cb_bf   = (unsigned short*)d_ws;                  // 4 MB
    float*          enorm05 = (float*)((char*)d_ws + (size_t)C_ * K_ * D_ * 2);

    float* out_z   = (float*)d_out;
    float* out_idx = out_z + (size_t)NTOK * D_;

    vq_prep<<<C_ * K_, 32, 0, stream>>>(cb, cb_bf, enorm05);

    (void)hipFuncSetAttribute((const void*)vq_main,
                              hipFuncAttributeMaxDynamicSharedMemorySize,
                              SMEM_TOTAL);
    vq_main<<<NTOK / 64, 128, SMEM_TOTAL, stream>>>(x_in, cb, cb_bf, enorm05,
                                                    out_z, out_idx);
}
